// DRMGCN_80900003987499
// MI455X (gfx1250) — matrix operand order repaired and checked by reading, not compiled, not run on hardware
//
#include <hip/hip_runtime.h>
#include <hip/hip_bf16.h>

// ---------------- constants from the reference ----------------
#define N_NODES 10000
#define NPAD    10016              // 313 * 32 : M padded for 32x32 wave tiles
#define FEAT    256
#define NEDGE   320000
#define NLAYERS 5
#define OCH     128
#define CATF    (NLAYERS * FEAT)   // 1280

typedef _Float16 h16;
typedef __attribute__((ext_vector_type(16))) _Float16 v16h;
typedef __attribute__((ext_vector_type(8)))  float    v8f;
typedef __attribute__((ext_vector_type(4)))  unsigned u32x4;

union Frag { u32x4 q[2]; v16h h; };

#define WMMA_F16(a, b, c) \
    __builtin_amdgcn_wmma_f32_16x16x32_f16(false, (a), false, (b), (short)0, (c), false, false)

// =====================================================================
// WMMA GEMM:  C[M,N] = A[M,K] @ Bt[N,K]^T (+bias), A/Bt row-major f16.
// 2x2 register blocking: each wave computes a 32x32 C block (4 WMMA
// per K=32 step from 4 fragment loads -> 2x the L2 arithmetic
// intensity of the unblocked version). 8 waves per 256-thread block.
// A rows must be allocated up to ceil32(M); stores masked per sub-tile
// (the 16-row sub-tile boundary aligns with M,N since M%16==N%16==0).
// =====================================================================
template<int ADD_BIAS, int OUT16>
__global__ __launch_bounds__(256) void wmma_gemm32(
    const h16* __restrict__ A, const h16* __restrict__ Bt,
    float* __restrict__ Cf, h16* __restrict__ Ch,
    const float* __restrict__ bias,
    int M, int N, int K, int lda, int ldb, int ldc)
{
    const int wave = threadIdx.x >> 5;
    const int lane = threadIdx.x & 31;
    const int half = lane >> 4;
    const int mr   = lane & 15;

    const int Nt2 = (N + 31) >> 5;
    const int Mt2 = (M + 31) >> 5;
    const int tile = blockIdx.x * 8 + wave;
    if (tile >= Mt2 * Nt2) return;
    const int tm2 = tile / Nt2;
    const int tn2 = tile % Nt2;

    const h16* arow0 = A + (size_t)(tm2 * 32 + mr) * lda;
    const h16* arow1 = arow0 + (size_t)16 * lda;
    const h16* brow0 = Bt + (size_t)(tn2 * 32 + mr) * ldb;
    const h16* brow1 = brow0 + (size_t)16 * ldb;

    v8f c00 = {}, c01 = {}, c10 = {}, c11 = {};
    for (int k0 = 0; k0 < K; k0 += 32) {
        Frag a0, a1, b0, b1;
        // A-fragment: lane half h covers K = {h*8..h*8+7} U {16+h*8..16+h*8+7}
        a0.q[0] = *(const u32x4*)(arow0 + k0 + half * 8);
        a0.q[1] = *(const u32x4*)(arow0 + k0 + 16 + half * 8);
        a1.q[0] = *(const u32x4*)(arow1 + k0 + half * 8);
        a1.q[1] = *(const u32x4*)(arow1 + k0 + 16 + half * 8);
        // B-fragment: lane half h covers K = {h*8..h*8+7} U {16+h*8..16+h*8+7}
        b0.q[0] = *(const u32x4*)(brow0 + k0 + half * 8);
        b0.q[1] = *(const u32x4*)(brow0 + k0 + 16 + half * 8);
        b1.q[0] = *(const u32x4*)(brow1 + k0 + half * 8);
        b1.q[1] = *(const u32x4*)(brow1 + k0 + 16 + half * 8);
        c00 = WMMA_F16(a0.h, b0.h, c00);
        c01 = WMMA_F16(a0.h, b1.h, c01);
        c10 = WMMA_F16(a1.h, b0.h, c10);
        c11 = WMMA_F16(a1.h, b1.h, c11);
    }

    auto store_tile = [&](const v8f& acc, int rbase, int cbase) {
        const int col = cbase + mr;
        const float bv = ADD_BIAS ? bias[col] : 0.0f;
#pragma unroll
        for (int r = 0; r < 8; ++r) {
            const int row = rbase + r + half * 8;   // C layout: VGPR r -> M = r + 8*half
            const float v = acc[r] + bv;
            if (OUT16) Ch[(size_t)row * ldc + col] = (h16)v;
            else       Cf[(size_t)row * ldc + col] = v;
        }
    };
    const int r0 = tm2 * 32, r1 = r0 + 16;
    const int c0 = tn2 * 32, c1 = c0 + 16;
    store_tile(c00, r0, c0);
    if (c1 < N)           store_tile(c01, r0, c1);
    if (r1 < M)           store_tile(c10, r1, c0);
    if (r1 < M && c1 < N) store_tile(c11, r1, c1);
}

// ---------------- graph / elementwise kernels ----------------
__global__ void k_deg_init(float* deg, unsigned* amax, int n) {
    int i = blockIdx.x * blockDim.x + threadIdx.x;
    if (i < n) deg[i] = 1.0f;                 // self-loop weight
    if (i < NLAYERS) amax[i] = 0u;            // relu outputs >= 0 -> bits monotonic
}

__global__ void k_deg_add(const int* __restrict__ dst, const float* __restrict__ ew,
                          float* deg, int e) {
    int i = blockIdx.x * blockDim.x + threadIdx.x;
    if (i < e) atomicAdd(&deg[dst[i]], ew[i]);
}

__global__ void k_dinv(float* deg, int n) {   // deg -> dinv in place
    int i = blockIdx.x * blockDim.x + threadIdx.x;
    if (i < n) { float d = deg[i]; deg[i] = d > 0.0f ? rsqrtf(d) : 0.0f; }
}

__global__ void k_norm(const int* __restrict__ src, const int* __restrict__ dst,
                       const float* __restrict__ ew, const float* __restrict__ dinv,
                       float* norm, int e) {
    int i = blockIdx.x * blockDim.x + threadIdx.x;
    if (i < e) norm[i] = dinv[src[i]] * ew[i] * dinv[dst[i]];
}

__global__ void k_f32_to_f16(const float* __restrict__ in, h16* __restrict__ out, long n) {
    long i = (long)blockIdx.x * blockDim.x + threadIdx.x;
    if (i < n) out[i] = (h16)in[i];
}

__global__ void k_wt_transpose(const float* __restrict__ W, h16* __restrict__ Wt) {
    int idx = blockIdx.x * blockDim.x + threadIdx.x;
    if (idx >= NLAYERS * FEAT * FEAT) return;
    int l  = idx / (FEAT * FEAT);
    int r  = idx % (FEAT * FEAT);
    int fo = r / FEAT, fi = r % FEAT;
    Wt[idx] = (h16)W[l * FEAT * FEAT + fi * FEAT + fo];
}

// agg[n,f] = dinv[n]^2 * H[n,f] + b[f]   (self-loop term + bias)
__global__ __launch_bounds__(256) void k_init_agg(const float* __restrict__ H,
                                                  const float* __restrict__ dinv,
                                                  const float* __restrict__ b,
                                                  float* __restrict__ agg) {
    int n = blockIdx.x, f = threadIdx.x;
    float dv = dinv[n];
    agg[(size_t)n * FEAT + f] = dv * dv * H[(size_t)n * FEAT + f] + b[f];
}

#define EPB 16
__global__ __launch_bounds__(256) void k_scatter(const int* __restrict__ src,
                                                 const int* __restrict__ dst,
                                                 const float* __restrict__ norm,
                                                 const float* __restrict__ H,
                                                 float* __restrict__ agg, int E) {
    __shared__ int   ss[EPB], sd[EPB];
    __shared__ float sn[EPB];
    const int t  = threadIdx.x;
    const int e0 = blockIdx.x * EPB;
    if (t < EPB)            { int e = e0 + t;           ss[t]           = (e < E) ? src[e]  : 0;    }
    else if (t < 2 * EPB)   { int e = e0 + t - EPB;     sd[t - EPB]     = (e < E) ? dst[e]  : 0;    }
    else if (t < 3 * EPB)   { int e = e0 + t - 2 * EPB; sn[t - 2 * EPB] = (e < E) ? norm[e] : 0.0f; }
    __syncthreads();
#pragma unroll
    for (int i = 0; i < EPB; ++i) {
        int e = e0 + i;
        if (e >= E) break;
        atomicAdd(&agg[(size_t)sd[i] * FEAT + t], sn[i] * H[(size_t)ss[i] * FEAT + t]);
    }
}

// relu, store f16 into stacked feats, track per-layer global max
__global__ __launch_bounds__(256) void k_relu_max(const float* __restrict__ agg,
                                                  h16* __restrict__ feats,
                                                  unsigned* __restrict__ amax, int layer) {
    __shared__ float red[256];
    int n = blockIdx.x, f = threadIdx.x;
    float v = agg[(size_t)n * FEAT + f];
    v = v > 0.0f ? v : 0.0f;
    feats[(size_t)n * CATF + layer * FEAT + f] = (h16)v;
    red[f] = v;
    __syncthreads();
    for (int s = 128; s > 0; s >>= 1) {
        if (f < s) red[f] = fmaxf(red[f], red[f + s]);
        __syncthreads();
    }
    if (f == 0) atomicMax(&amax[layer], __float_as_uint(red[0]));
}

// tiny attention MLP: att = sigmoid(fc2(relu(fc1(max))))
__global__ void k_att(const unsigned* __restrict__ amax,
                      const float* __restrict__ fc1w, const float* __restrict__ fc1b,
                      const float* __restrict__ fc2w, const float* __restrict__ fc2b,
                      float* __restrict__ att) {
    if (threadIdx.x != 0 || blockIdx.x != 0) return;
    float m[NLAYERS], a1[5 * NLAYERS];
    for (int c = 0; c < NLAYERS; ++c) m[c] = __uint_as_float(amax[c]);
    for (int j = 0; j < 5 * NLAYERS; ++j) {
        float s = fc1b[j];
        for (int c = 0; c < NLAYERS; ++c) s += fc1w[j * NLAYERS + c] * m[c];
        a1[j] = fmaxf(s, 0.0f);
    }
    for (int c = 0; c < NLAYERS; ++c) {
        float s = fc2b[c];
        for (int j = 0; j < 5 * NLAYERS; ++j) s += fc2w[c * 5 * NLAYERS + j] * a1[j];
        att[c] = 1.0f / (1.0f + __expf(-s));
    }
}

// Bt_conv[o, c*256+f] = att[c] * cw[o,c,f]   (relu(att*X)==att*X since X>=0, att>0)
__global__ void k_btconv(const float* __restrict__ cw, const float* __restrict__ att,
                         h16* __restrict__ Bt) {
    int idx = blockIdx.x * blockDim.x + threadIdx.x;
    if (idx >= OCH * CATF) return;
    int c = (idx % CATF) / FEAT;
    Bt[idx] = (h16)(att[c] * cw[idx]);
}

static inline int gemm_blocks(int M, int N) {
    int tiles = ((M + 31) / 32) * ((N + 31) / 32);
    return (tiles + 7) / 8;
}

// ---------------- host-side branch driver ----------------
static void run_branch(const float* x0, const float* W, const float* b, const float* ew,
                       const int* src, const int* dst,
                       const float* fc1w, const float* fc1b,
                       const float* fc2w, const float* fc2b,
                       const float* cw, const float* cb,
                       float* deg, float* norm, h16* x0f16, h16* feats,
                       float* H, float* agg, h16* Wt, h16* Btc,
                       unsigned* amax, float* att, h16* out16, hipStream_t s)
{
    k_wt_transpose<<<(NLAYERS * FEAT * FEAT + 255) / 256, 256, 0, s>>>(W, Wt);
    k_f32_to_f16<<<(N_NODES * FEAT + 255) / 256, 256, 0, s>>>(x0, x0f16, (long)N_NODES * FEAT);
    k_deg_init<<<(N_NODES + 255) / 256, 256, 0, s>>>(deg, amax, N_NODES);
    k_deg_add<<<(NEDGE + 255) / 256, 256, 0, s>>>(dst, ew, deg, NEDGE);
    k_dinv<<<(N_NODES + 255) / 256, 256, 0, s>>>(deg, N_NODES);
    k_norm<<<(NEDGE + 255) / 256, 256, 0, s>>>(src, dst, ew, deg, norm, NEDGE);

    for (int l = 0; l < NLAYERS; ++l) {
        const h16* Ain = (l == 0) ? x0f16 : (feats + (size_t)(l - 1) * FEAT);
        const int  lda = (l == 0) ? FEAT : CATF;
        wmma_gemm32<0, 0><<<gemm_blocks(N_NODES, FEAT), 256, 0, s>>>(
            Ain, Wt + (size_t)l * FEAT * FEAT, H, nullptr, nullptr,
            N_NODES, FEAT, FEAT, lda, FEAT, FEAT);
        k_init_agg<<<N_NODES, 256, 0, s>>>(H, deg, b + (size_t)l * FEAT, agg);
        k_scatter<<<(NEDGE + EPB - 1) / EPB, 256, 0, s>>>(src, dst, norm, H, agg, NEDGE);
        k_relu_max<<<N_NODES, 256, 0, s>>>(agg, feats, amax, l);
    }

    k_att<<<1, 32, 0, s>>>(amax, fc1w, fc1b, fc2w, fc2b, att);
    k_btconv<<<(OCH * CATF + 255) / 256, 256, 0, s>>>(cw, att, Btc);
    wmma_gemm32<1, 1><<<gemm_blocks(N_NODES, OCH), 256, 0, s>>>(
        feats, Btc, nullptr, out16, cb, N_NODES, OCH, CATF, CATF, CATF, OCH);
}

extern "C" void kernel_launch(void* const* d_in, const int* in_sizes, int n_in,
                              void* d_out, int out_size, void* d_ws, size_t ws_size,
                              hipStream_t stream)
{
    const float* x_m    = (const float*)d_in[0];
    const float* x_d    = (const float*)d_in[1];
    const float* w_m    = (const float*)d_in[2];
    const float* w_d    = (const float*)d_in[3];
    const float* Wx     = (const float*)d_in[4];
    const float* bx     = (const float*)d_in[5];
    const float* Wy     = (const float*)d_in[6];
    const float* by     = (const float*)d_in[7];
    const float* fc1x_w = (const float*)d_in[8];
    const float* fc1x_b = (const float*)d_in[9];
    const float* fc2x_w = (const float*)d_in[10];
    const float* fc2x_b = (const float*)d_in[11];
    const float* fc1y_w = (const float*)d_in[12];
    const float* fc1y_b = (const float*)d_in[13];
    const float* fc2y_w = (const float*)d_in[14];
    const float* fc2y_b = (const float*)d_in[15];
    const float* cnnx_w = (const float*)d_in[16];
    const float* cnnx_b = (const float*)d_in[17];
    const float* cnny_w = (const float*)d_in[18];
    const float* cnny_b = (const float*)d_in[19];
    const int*   em     = (const int*)d_in[20];   // [2,E]: src, dst
    const int*   ed     = (const int*)d_in[21];

    // workspace carve-up (~59 MB total), 256B aligned.
    // f16 node matrices are allocated with NPAD (=10016) rows so 32x32
    // wave tiles may read past row 9999; those accumulators are discarded.
    char* w = (char*)d_ws; size_t off = 0;
    auto carve = [&](size_t bytes) -> void* {
        void* p = (void*)(w + off); off += (bytes + 255) & ~(size_t)255; return p;
    };
    float*    deg   = (float*)   carve((size_t)N_NODES * 4);
    float*    norm  = (float*)   carve((size_t)NEDGE * 4);
    h16*      x0f16 = (h16*)     carve((size_t)NPAD * FEAT * 2);
    h16*      feats = (h16*)     carve((size_t)NPAD * CATF * 2);
    float*    H     = (float*)   carve((size_t)N_NODES * FEAT * 4);
    float*    agg   = (float*)   carve((size_t)N_NODES * FEAT * 4);
    h16*      Wt    = (h16*)     carve((size_t)NLAYERS * FEAT * FEAT * 2);
    h16*      Btc   = (h16*)     carve((size_t)OCH * CATF * 2);
    unsigned* amax  = (unsigned*)carve(64);
    float*    att   = (float*)   carve(64);
    h16*      xo    = (h16*)     carve((size_t)NPAD * OCH * 2);
    h16*      yo    = (h16*)     carve((size_t)NPAD * OCH * 2);

    run_branch(x_m, Wx, bx, w_m, em, em + NEDGE,
               fc1x_w, fc1x_b, fc2x_w, fc2x_b, cnnx_w, cnnx_b,
               deg, norm, x0f16, feats, H, agg, Wt, Btc, amax, att, xo, stream);

    run_branch(x_d, Wy, by, w_d, ed, ed + NEDGE,
               fc1y_w, fc1y_b, fc2y_w, fc2y_b, cnny_w, cnny_b,
               deg, norm, x0f16, feats, H, agg, Wt, Btc, amax, att, yo, stream);

    // final: out[i,j] = sum_k xo[i,k] * yo[j,k]   (A @ Bt^T with Bt = yo)
    wmma_gemm32<0, 0><<<gemm_blocks(N_NODES, N_NODES), 256, 0, stream>>>(
        xo, yo, (float*)d_out, nullptr, nullptr,
        N_NODES, N_NODES, OCH, OCH, OCH, N_NODES);
}
